// WindowAttention3D_75909251990229
// MI455X (gfx1250) — matrix operand order repaired and checked by reading, not compiled, not run on hardware
//
#include <hip/hip_runtime.h>
#include <hip/hip_bf16.h>

typedef _Float16 f16;
typedef _Float16 v16h __attribute__((ext_vector_type(16)));
typedef _Float16 v8h  __attribute__((ext_vector_type(8)));
typedef float    v8f  __attribute__((ext_vector_type(8)));

// ---------------------------------------------------------------------------
// WMMA helpers (V_WMMA_F32_16X16X32_F16, wave32)
// D/C layout: elem r -> row M = r + 8*(lane/16), col N = lane%16
// A layout:   lane holds row M=lane%16; elem e -> K = 16*(lane/16) + e
// B layout:   lane holds col N=lane%16; elem e -> K = 16*(lane/16) + e
// ---------------------------------------------------------------------------
#define WMMA_F16(a, b, c) \
  __builtin_amdgcn_wmma_f32_16x16x32_f16(false, (a), false, (b), (short)0, (c), false, false)

__device__ __forceinline__ v16h frag_cat(v8h lo, v8h hi) {
  v16h r;
#pragma unroll
  for (int i = 0; i < 8; ++i) { r[i] = lo[i]; r[i + 8] = hi[i]; }
  return r;
}

__device__ __forceinline__ v8f vzero8() {
  v8f z = {0.f, 0.f, 0.f, 0.f, 0.f, 0.f, 0.f, 0.f};
  return z;
}

__device__ __forceinline__ v8h hzero8() {
  v8h z;
#pragma unroll
  for (int i = 0; i < 8; ++i) z[i] = (f16)0.f;
  return z;
}

// A fragment from an LDS tile stored row-major [rows][LDA halfs]
template <int LDA>
__device__ __forceinline__ v16h load_a_frag(const f16* s, int row0, int kbase) {
  const int lane = threadIdx.x & 31;
  const int half = lane >> 4;
  const int m = lane & 15;
  const f16* p = s + (row0 + m) * LDA + kbase;
  v8h lo = *(const v8h*)(p + 16 * half);
  v8h hi = *(const v8h*)(p + 16 * half + 8);
  return frag_cat(lo, hi);
}

// B fragment from LDS tile stored as Bt[n][k]
template <int LDB>
__device__ __forceinline__ v16h load_b_frag(const f16* s, int n0, int kbase) {
  const int lane = threadIdx.x & 31;
  const int half = lane >> 4;
  const int n = lane & 15;
  const f16* p = s + (n0 + n) * LDB + kbase + 16 * half;
  v8h lo = *(const v8h*)(p);
  v8h hi = *(const v8h*)(p + 8);
  return frag_cat(lo, hi);
}

// ---------------------------------------------------------------------------
// Kernel 1: qkv = x @ qkv_w.T   (f32 in -> f16 out), M=100352, N=1536, K=512
// ---------------------------------------------------------------------------
#define LDT 40  // padded LDS row stride (halfs) for 32-wide K slices

__global__ __launch_bounds__(256) void qkv_gemm_kernel(
    const float* __restrict__ X, const float* __restrict__ W,
    f16* __restrict__ Out, int M, int N, int K) {
  __shared__ __align__(16) f16 As[128 * LDT];
  __shared__ __align__(16) f16 Bs[128 * LDT];

  const int tid = threadIdx.x;
  const int wave = tid >> 5;
  const int lane = tid & 31;
  const int half = lane >> 4;
  const int nn = lane & 15;
  const int m0 = blockIdx.x * 128;
  const int n0 = blockIdx.y * 128;

  v8f acc[8];
#pragma unroll
  for (int j = 0; j < 8; ++j) acc[j] = vzero8();

  const int lrow = tid >> 1;        // 0..127
  const int lcol = (tid & 1) * 16;  // 0 or 16

  const float* xp = X + (size_t)(m0 + lrow) * K + lcol;
  const float* wp = W + (size_t)(n0 + lrow) * K + lcol;

  for (int k0 = 0; k0 < K; k0 += 32) {
#pragma unroll
    for (int i = 0; i < 16; i += 4) {
      float4 xa = *(const float4*)(xp + k0 + i);
      float4 wa = *(const float4*)(wp + k0 + i);
      f16* ap = &As[lrow * LDT + lcol + i];
      ap[0] = (f16)xa.x; ap[1] = (f16)xa.y; ap[2] = (f16)xa.z; ap[3] = (f16)xa.w;
      f16* bp = &Bs[lrow * LDT + lcol + i];
      bp[0] = (f16)wa.x; bp[1] = (f16)wa.y; bp[2] = (f16)wa.z; bp[3] = (f16)wa.w;
    }
    if (k0 + 32 < K) {  // global_prefetch_b8 for next K slice
      __builtin_prefetch(xp + k0 + 32, 0, 0);
      __builtin_prefetch(wp + k0 + 32, 0, 0);
    }
    __syncthreads();

    v16h afrag = load_a_frag<LDT>(As, wave * 16, 0);
#pragma unroll
    for (int j0 = 0; j0 < 8; j0 += 4) {  // batch B frags to overlap DS with WMMA
      v16h bf[4];
#pragma unroll
      for (int j = 0; j < 4; ++j) bf[j] = load_b_frag<LDT>(Bs, (j0 + j) * 16, 0);
#pragma unroll
      for (int j = 0; j < 4; ++j) acc[j0 + j] = WMMA_F16(afrag, bf[j], acc[j0 + j]);
    }
    __syncthreads();
  }

#pragma unroll
  for (int j = 0; j < 8; ++j) {
    const int col = n0 + j * 16 + nn;
#pragma unroll
    for (int r = 0; r < 8; ++r) {
      const int row = m0 + wave * 16 + r + 8 * half;
      Out[(size_t)row * N + col] = (f16)acc[j][r];
    }
  }
}

// ---------------------------------------------------------------------------
// Kernel 2: per-(batch,head) attention. N=98 padded to 112, d=32.
// 224 threads = 7 waves; wave w owns query strip [w*16, w*16+16).
// ---------------------------------------------------------------------------
#define AT_LDQ 40    // q/k LDS stride (32 K + pad)
#define AT_LDS 116   // score f32 stride (29*16B rows -> float4 aligned)
#define AT_LDP 136   // P / Vt stride (128 K + pad)

__global__ __launch_bounds__(224) void attn_kernel(
    const f16* __restrict__ QKV, const float* __restrict__ bias_table,
    const int* __restrict__ rel_index, f16* __restrict__ Out) {
  __shared__ __align__(16) f16 qs[112 * AT_LDQ];
  __shared__ __align__(16) f16 ks[112 * AT_LDQ];
  __shared__ __align__(16) f16 vt[32 * AT_LDP];    // transposed V: [d][key]
  __shared__ __align__(16) float S[112 * AT_LDS];  // scores f32
  __shared__ __align__(16) f16 P[112 * AT_LDP];    // probs f16, K padded to 128

  const int b = blockIdx.x >> 4;
  const int h = blockIdx.x & 15;
  const int tid = threadIdx.x;
  const int wave = tid >> 5;
  const int lane = tid & 31;
  const int half = lane >> 4;
  const int nn = lane & 15;
  const float scale = 0.1767766952966369f;  // 32^-0.5

  const size_t base = (size_t)b * 98 * 1536 + (size_t)h * 32;

  // q (scaled) and k: vectorized 16B copies into LDS
  for (int idx = tid; idx < 98 * 4; idx += 224) {
    const int row = idx >> 2;
    const int c8 = (idx & 3) * 8;
    const f16* p = QKV + base + (size_t)row * 1536 + c8;
    v8h qv = *(const v8h*)(p);
    v8h kv = *(const v8h*)(p + 512);
#pragma unroll
    for (int i = 0; i < 8; ++i) qv[i] = (f16)((float)qv[i] * scale);
    *(v8h*)&qs[row * AT_LDQ + c8] = qv;
    *(v8h*)&ks[row * AT_LDQ + c8] = kv;
  }
  // v transposed scatter
  for (int idx = tid; idx < 98 * 32; idx += 224) {
    const int row = idx >> 5;
    const int col = idx & 31;
    vt[col * AT_LDP + row] = QKV[base + (size_t)row * 1536 + 1024 + col];
  }
  // zero padded token rows of q/k (98..111)
  for (int idx = tid; idx < 14 * 4; idx += 224) {
    const int row = 98 + (idx >> 2);
    const int c8 = (idx & 3) * 8;
    v8h z = hzero8();
    *(v8h*)&qs[row * AT_LDQ + c8] = z;
    *(v8h*)&ks[row * AT_LDQ + c8] = z;
  }
  // zero padded key columns of vt (98..127) and P (112..127)
  for (int idx = tid; idx < 32 * 30; idx += 224)
    vt[(idx / 30) * AT_LDP + 98 + (idx % 30)] = (f16)0.f;
  for (int idx = tid; idx < 112 * 16; idx += 224)
    P[(idx >> 4) * AT_LDP + 112 + (idx & 15)] = (f16)0.f;
  __syncthreads();

  // S = q @ k^T  (K=32), bias gathered at accumulator writeback
  {
    v16h aq = load_a_frag<AT_LDQ>(qs, wave * 16, 0);
    v16h bk[7];
#pragma unroll
    for (int j = 0; j < 7; ++j) bk[j] = load_b_frag<AT_LDQ>(ks, j * 16, 0);
    v8f acc[7];
#pragma unroll
    for (int j = 0; j < 7; ++j) acc[j] = vzero8();
#pragma unroll
    for (int j = 0; j < 7; ++j) acc[j] = WMMA_F16(aq, bk[j], acc[j]);

#pragma unroll
    for (int j = 0; j < 7; ++j) {
      const int m = j * 16 + nn;  // key index
#pragma unroll
      for (int r = 0; r < 8; ++r) {
        const int n = wave * 16 + r + 8 * half;  // query index
        float v = acc[j][r];
        if (n < 98 && m < 98)
          v += bias_table[rel_index[n * 98 + m] * 16 + h];
        else
          v = -1e30f;  // mask padding
        S[n * AT_LDS + m] = v;
      }
    }
  }
  __syncthreads();

  // row softmax, float4-vectorized, one thread per query row
  if (tid < 112) {
    float4* row4 = (float4*)&S[tid * AT_LDS];
    float mx = -3.0e38f;
#pragma unroll 4
    for (int m = 0; m < 28; ++m) {
      float4 v = row4[m];
      mx = fmaxf(mx, fmaxf(fmaxf(v.x, v.y), fmaxf(v.z, v.w)));
    }
    float sum = 0.f;
#pragma unroll 4
    for (int m = 0; m < 28; ++m) {
      float4 v = row4[m];
      v.x = __expf(v.x - mx); v.y = __expf(v.y - mx);
      v.z = __expf(v.z - mx); v.w = __expf(v.w - mx);
      sum += (v.x + v.y) + (v.z + v.w);
      row4[m] = v;
    }
    const float inv = 1.f / sum;
    f16* prow = &P[tid * AT_LDP];
#pragma unroll 4
    for (int m = 0; m < 28; ++m) {
      float4 v = row4[m];
      prow[m * 4 + 0] = (f16)(v.x * inv);
      prow[m * 4 + 1] = (f16)(v.y * inv);
      prow[m * 4 + 2] = (f16)(v.z * inv);
      prow[m * 4 + 3] = (f16)(v.w * inv);
    }
  }
  __syncthreads();

  // O = P @ V   (K=128 in 4 steps, 2 subtiles over d=32)
  {
    v8f o[2] = {vzero8(), vzero8()};
#pragma unroll
    for (int kk = 0; kk < 4; ++kk) {
      v16h ap = load_a_frag<AT_LDP>(P, wave * 16, kk * 32);
      v16h bv0 = load_b_frag<AT_LDP>(vt, 0, kk * 32);
      v16h bv1 = load_b_frag<AT_LDP>(vt, 16, kk * 32);
      o[0] = WMMA_F16(ap, bv0, o[0]);
      o[1] = WMMA_F16(ap, bv1, o[1]);
    }
#pragma unroll
    for (int j = 0; j < 2; ++j) {
      const int dd = j * 16 + nn;
#pragma unroll
      for (int r = 0; r < 8; ++r) {
        const int n = wave * 16 + r + 8 * half;
        if (n < 98)
          Out[((size_t)b * 98 + n) * 512 + h * 32 + dd] = (f16)o[j][r];
      }
    }
  }
}

// ---------------------------------------------------------------------------
// Kernel 3: out = attn @ proj_w.T + proj_b   (f16 A via async-to-LDS copy)
// ---------------------------------------------------------------------------
__global__ __launch_bounds__(256) void proj_gemm_kernel(
    const f16* __restrict__ Ah, const float* __restrict__ W,
    const float* __restrict__ bias, float* __restrict__ Out,
    int M, int N, int K) {
  __shared__ __align__(16) f16 As[128 * LDT];
  __shared__ __align__(16) f16 Bs[128 * LDT];

  const int tid = threadIdx.x;
  const int wave = tid >> 5;
  const int lane = tid & 31;
  const int half = lane >> 4;
  const int nn = lane & 15;
  const int m0 = blockIdx.x * 128;
  const int n0 = blockIdx.y * 128;

  v8f acc[8];
#pragma unroll
  for (int j = 0; j < 8; ++j) acc[j] = vzero8();

  const int lrow = tid >> 1;
  const int lcol = (tid & 1) * 16;
  const float* wp = W + (size_t)(n0 + lrow) * K + lcol;

  // LDS byte offset of As for async-to-LDS destination (low 32 bits of generic ptr)
  const unsigned as_base = (unsigned)(unsigned long long)(uintptr_t)(&As[0]);
  // each thread moves two 16B chunks of the A tile: ids tid and tid+256
  const int arow0 = tid >> 2, achk0 = tid & 3;          // ids 0..255
  const int arow1 = (tid + 256) >> 2, achk1 = tid & 3;  // ids 256..511

  for (int k0 = 0; k0 < K; k0 += 32) {
    // A tile: pure f16 copy -> async DMA to LDS (ASYNCcnt-tracked)
    {
      const f16* g0 = Ah + (size_t)(m0 + arow0) * K + k0 + achk0 * 8;
      const f16* g1 = Ah + (size_t)(m0 + arow1) * K + k0 + achk1 * 8;
      const unsigned l0 = as_base + arow0 * (LDT * 2) + achk0 * 16;
      const unsigned l1 = as_base + arow1 * (LDT * 2) + achk1 * 16;
      asm volatile("global_load_async_to_lds_b128 %0, %1, off"
                   :: "v"(l0), "v"(g0) : "memory");
      asm volatile("global_load_async_to_lds_b128 %0, %1, off"
                   :: "v"(l1), "v"(g1) : "memory");
    }
    // B tile: f32 -> f16 convert through VGPRs
#pragma unroll
    for (int i = 0; i < 16; i += 4) {
      float4 wa = *(const float4*)(wp + k0 + i);
      f16* bp = &Bs[lrow * LDT + lcol + i];
      bp[0] = (f16)wa.x; bp[1] = (f16)wa.y; bp[2] = (f16)wa.z; bp[3] = (f16)wa.w;
    }
    if (k0 + 32 < K) __builtin_prefetch(wp + k0 + 32, 0, 0);
    asm volatile("s_wait_asynccnt 0x0" ::: "memory");
    __syncthreads();

    v16h afrag = load_a_frag<LDT>(As, wave * 16, 0);
#pragma unroll
    for (int j0 = 0; j0 < 8; j0 += 4) {
      v16h bf[4];
#pragma unroll
      for (int j = 0; j < 4; ++j) bf[j] = load_b_frag<LDT>(Bs, (j0 + j) * 16, 0);
#pragma unroll
      for (int j = 0; j < 4; ++j) acc[j0 + j] = WMMA_F16(afrag, bf[j], acc[j0 + j]);
    }
    __syncthreads();
  }

#pragma unroll
  for (int j = 0; j < 8; ++j) {
    const int col = n0 + j * 16 + nn;
    const float bv = bias[col];
#pragma unroll
    for (int r = 0; r < 8; ++r) {
      const int row = m0 + wave * 16 + r + 8 * half;
      Out[(size_t)row * N + col] = acc[j][r] + bv;
    }
  }
}

// ---------------------------------------------------------------------------
extern "C" void kernel_launch(void* const* d_in, const int* in_sizes, int n_in,
                              void* d_out, int out_size, void* d_ws, size_t ws_size,
                              hipStream_t stream) {
  const float* x          = (const float*)d_in[0];  // [1024, 98, 512]
  const float* qkv_w      = (const float*)d_in[1];  // [1536, 512]
  const float* proj_w     = (const float*)d_in[2];  // [512, 512]
  const float* proj_b     = (const float*)d_in[3];  // [512]
  const float* bias_table = (const float*)d_in[4];  // [507, 16]
  const int*   rel_index  = (const int*)d_in[5];    // [98, 98]
  float* out = (float*)d_out;

  const int M = 1024 * 98;  // 100352 (divisible by 128)
  // workspace: qkv f16 [M,1536] (308 MB) then attn-out f16 [M,512] (103 MB)
  f16* qkvh  = (f16*)d_ws;
  f16* attnh = qkvh + (size_t)M * 1536;

  qkv_gemm_kernel<<<dim3(M / 128, 1536 / 128), 256, 0, stream>>>(
      x, qkv_w, qkvh, M, 1536, 512);
  attn_kernel<<<1024 * 16, 224, 0, stream>>>(qkvh, bias_table, rel_index, attnh);
  proj_gemm_kernel<<<dim3(M / 128, 512 / 128), 256, 0, stream>>>(
      attnh, proj_w, proj_b, out, M, 512, 512);
}
